// ReCausalSelfAttentionL3_90340342104845
// MI455X (gfx1250) — compile-verified, matrix operand order checked by reading, not run on hardware
//
#include <hip/hip_runtime.h>
#include <stdint.h>

// Problem constants (from reference)
#define B_    8
#define T_    1024
#define C_    1024
#define H_    16
#define HD_   64
#define NU_   128
#define M_    16
#define LKV_  (T_ + 2)
#define LKVP_ 1056          // padded key stride for transposed V (mult of 32)
#define SCALE_ 0.125f

typedef float  v8f   __attribute__((ext_vector_type(8)));
typedef __bf16 v16bf __attribute__((ext_vector_type(16)));
typedef __bf16 v8bf  __attribute__((ext_vector_type(8)));

__device__ inline __bf16 f2bf(float f) {
  uint32_t u; __builtin_memcpy(&u, &f, 4);
  uint32_t r = u + 0x7FFFu + ((u >> 16) & 1u);   // round-to-nearest-even
  uint16_t h = (uint16_t)(r >> 16);
  __bf16 o; __builtin_memcpy(&o, &h, 2);
  return o;
}

// Async global->LDS copy of 16B per lane (CDNA5 TDM-adjacent path, ASYNCcnt).
// LDS offset = low 32 bits of the generic pointer (ISA 10.2: LDS aperture uses
// addr[31:0] as the LDS byte address).
__device__ inline void async_copy16(const __bf16* lds_dst, const __bf16* gsrc) {
  unsigned lds_off = (unsigned)(uintptr_t)lds_dst;
  asm volatile("global_load_async_to_lds_b128 %0, %1, off"
               :: "v"(lds_off), "v"(gsrc)
               : "memory");
}
__device__ inline void async_wait0() {
  asm volatile("s_wait_asynccnt 0x0" ::: "memory");
}

// Build a 16-half A/B fragment from two 8-element contiguous runs.
// ISA layout (16-bit A 16x32 / transposed B): lanes 0-15 hold K {0..7,16..23},
// lanes 16-31 hold K {8..15,24..31}; caller passes off = (lane&16)?8:0.
__device__ inline v16bf make_frag(const __bf16* p0, const __bf16* p1) {
  v8bf a = *(const v8bf*)p0;
  v8bf b = *(const v8bf*)p1;
  v16bf r;
#pragma unroll
  for (int i = 0; i < 8; ++i) { r[i] = a[i]; r[i + 8] = b[i]; }
  return r;
}

__device__ inline v8f wmma_bf16(v16bf a, v16bf b, v8f c) {
  return __builtin_amdgcn_wmma_f32_16x16x32_bf16(false, a, false, b, (short)0, c,
                                                 false, false);
}

// ------------------------------------------------- weight transpose + convert
// src [K][N] f32 -> dst [N][K] bf16  (C_ x C_), 32x32 LDS-tiled
__global__ __launch_bounds__(256)
void transpose_cvt(const float* __restrict__ src, __bf16* __restrict__ dst) {
  __shared__ float tile[32][33];
  const int tx = threadIdx.x & 31, ty = threadIdx.x >> 5;
  const int k0 = blockIdx.x * 32, n0 = blockIdx.y * 32;
#pragma unroll
  for (int p = 0; p < 4; ++p)
    tile[ty + p * 8][tx] = src[(size_t)(k0 + ty + p * 8) * C_ + n0 + tx];
  __syncthreads();
#pragma unroll
  for (int p = 0; p < 4; ++p)
    dst[(size_t)(n0 + ty + p * 8) * C_ + k0 + tx] = f2bf(tile[tx][ty + p * 8]);
}

// out[row,:] = A[row,:] @ W + bias   (fp32, tiny: used for qm/km projections)
__global__ __launch_bounds__(256)
void row_proj(const float* __restrict__ A, const float* __restrict__ W,
              const float* __restrict__ bias, float* __restrict__ out) {
  __shared__ float arow[C_];
  const int row = blockIdx.y;
  const int col = blockIdx.x * 256 + threadIdx.x;
  const float* ap = A + (size_t)row * C_;
  for (int i = threadIdx.x; i < C_; i += 256) arow[i] = ap[i];
  __syncthreads();
  float sum = bias[col];
  for (int kk = 0; kk < C_; ++kk) sum = fmaf(arow[kk], W[(size_t)kk * C_ + col], sum);
  out[(size_t)row * C_ + col] = sum;
}

// memory-compression attention: one wave per (b,h); M=16 keys, HD=64
__global__ __launch_bounds__(256)
void mem_attn(const float* __restrict__ qm, const float* __restrict__ km,
              float* __restrict__ mem) {
  const int gw = (blockIdx.x * blockDim.x + threadIdx.x) >> 5;
  const int lane = threadIdx.x & 31;
  const int b = gw / H_, h = gw % H_;
  const float* qh = qm + (size_t)b * C_ + h * HD_;
  const float* kh = km + (size_t)b * M_ * C_ + h * HD_;
  float sj = -1e30f;
  if (lane < M_) {
    const float* kr = kh + (size_t)lane * C_;
    float a = 0.f;
    for (int d = 0; d < HD_; ++d) a = fmaf(qh[d], kr[d], a);
    sj = a * SCALE_;
  }
  float mx = sj;
  for (int d = 1; d < 32; d <<= 1) mx = fmaxf(mx, __shfl_xor(mx, d, 32));
  float e = (lane < M_) ? __expf(sj - mx) : 0.f;
  float se = e;
  for (int d = 1; d < 32; d <<= 1) se += __shfl_xor(se, d, 32);
  const float a = e / se;
  float o0 = 0.f, o1 = 0.f;
  for (int j = 0; j < M_; ++j) {
    const float aj = __shfl(a, j, 32);
    o0 = fmaf(aj, kh[(size_t)j * C_ + lane], o0);
    o1 = fmaf(aj, kh[(size_t)j * C_ + lane + 32], o1);
  }
  mem[(size_t)b * C_ + h * HD_ + lane]      = o0;
  mem[(size_t)b * C_ + h * HD_ + lane + 32] = o1;
}

// cm = concat([mem, prev_m, x], seq dim) converted to bf16
__global__ __launch_bounds__(256)
void build_cm(const float* __restrict__ x, const float* __restrict__ prev_m,
              const float* __restrict__ mem, __bf16* __restrict__ cm) {
  const size_t n = (size_t)B_ * LKV_ * C_;
  size_t i = (size_t)blockIdx.x * 256 + threadIdx.x;
  if (i >= n) return;
  const int col = (int)(i % C_);
  const size_t rr = i / C_;
  const int row = (int)(rr % LKV_);
  const int b   = (int)(rr / LKV_);
  float val;
  if (row == 0)      val = mem[(size_t)b * C_ + col];
  else if (row == 1) val = prev_m[(size_t)b * C_ + col];
  else               val = x[((size_t)b * T_ + (row - 2)) * C_ + col];
  cm[i] = f2bf(val);
}

// -------------------------------------------------------------- bf16 WMMA GEMM
// Out[bb, row, :] = A[bb, row+off, :] @ Wt[N][K]^T + bias ; 128x128 tile, BK=32
// out_mode: 0 = f32 row-major, 1 = bf16 row-major, 2 = bf16 head-transposed
//           (vt[(bb*H + col/64)*64 + col%64][row], key-stride LKVP_)
// Staging uses GLOBAL_LOAD_ASYNC_TO_LDS_B128; OOB A rows are never issued, so
// their (stale) LDS rows pollute only accumulator rows that are never stored.
__global__ __launch_bounds__(256)
void gemm_bf16(const __bf16* __restrict__ A, int a_rows_pb, int a_row_off,
               const __bf16* __restrict__ Wt, const float* __restrict__ bias,
               void* __restrict__ Out, int out_rows_pb, int m_pb, int out_mode) {
  const int K = C_, N = C_;
  __shared__ __bf16 As[128][40];   // padded stride: conflict-free fragment reads
  __shared__ __bf16 Bs[128][40];   // Bs[n][k] (weights already [N][K])
  const int tid  = threadIdx.x;
  const int wave = tid >> 5, lane = tid & 31;
  const int lr    = lane & 15;
  const int off   = (lane & 16) ? 8 : 0;
  const int rbase = (lane & 16) ? 8 : 0;
  const int bM = blockIdx.x * 128;
  const int bN = blockIdx.y * 128;
  const int bb = blockIdx.z;
  const __bf16* Ab = A + ((size_t)bb * a_rows_pb + a_row_off) * K;
  const int wrow = (wave >> 1) * 32;
  const int wcol = (wave & 1) * 64;
  v8f acc[2][4];
#pragma unroll
  for (int mi = 0; mi < 2; ++mi)
#pragma unroll
    for (int ni = 0; ni < 4; ++ni)
#pragma unroll
      for (int i = 0; i < 8; ++i) acc[mi][ni][i] = 0.f;

  // A and B tiles are both 128 rows x 32 k: identical staging mapping
  const int srow = tid >> 1, skoff = (tid & 1) * 16;
  const bool avalid = (bM + srow) < m_pb;
  const __bf16* pa = Ab + (size_t)(bM + srow) * K + skoff;
  const __bf16* pw = Wt + (size_t)(bN + srow) * K + skoff;
  __bf16* ldsA = &As[srow][skoff];
  __bf16* ldsB = &Bs[srow][skoff];

  for (int k0 = 0; k0 < K; k0 += 32) {
    __syncthreads();                      // previous tile fully consumed
    if (avalid) {
      async_copy16(ldsA,     pa);
      async_copy16(ldsA + 8, pa + 8);
    }
    async_copy16(ldsB,     pw);
    async_copy16(ldsB + 8, pw + 8);
    if (k0 + 32 < K) {
      if (avalid) __builtin_prefetch(pa + 32, 0, 0);   // global_prefetch_b8
      __builtin_prefetch(pw + 32, 0, 0);
    }
    async_wait0();                        // this wave's async copies landed
    __syncthreads();                      // all waves' tiles visible

    v16bf af[2], bfr[4];
#pragma unroll
    for (int mi = 0; mi < 2; ++mi) {
      const __bf16* p = &As[wrow + mi * 16 + lr][0];
      af[mi] = make_frag(p + off, p + off + 16);
    }
#pragma unroll
    for (int ni = 0; ni < 4; ++ni) {
      const __bf16* p = &Bs[wcol + ni * 16 + lr][0];
      bfr[ni] = make_frag(p + off, p + off + 16);
    }
#pragma unroll
    for (int mi = 0; mi < 2; ++mi)
#pragma unroll
      for (int ni = 0; ni < 4; ++ni)
        acc[mi][ni] = wmma_bf16(af[mi], bfr[ni], acc[mi][ni]);

    pa += 32;
    pw += 32;
  }

#pragma unroll
  for (int mi = 0; mi < 2; ++mi)
#pragma unroll
    for (int ni = 0; ni < 4; ++ni) {
      const int col = bN + wcol + ni * 16 + lr;
      const float bv = bias ? bias[col] : 0.f;
#pragma unroll
      for (int i = 0; i < 8; ++i) {
        const int row = bM + wrow + mi * 16 + rbase + i;
        if (row < m_pb) {
          const float val = acc[mi][ni][i] + bv;
          if (out_mode == 0) {
            ((float*)Out)[((size_t)bb * out_rows_pb + row) * (size_t)N + col] = val;
          } else if (out_mode == 1) {
            ((__bf16*)Out)[((size_t)bb * out_rows_pb + row) * (size_t)N + col] = f2bf(val);
          } else {
            const int hh = col >> 6, hd = col & 63;
            ((__bf16*)Out)[(((size_t)bb * H_ + hh) * HD_ + hd) * LKVP_ + row] = f2bf(val);
          }
        }
      }
    }
}

// ----------------------------------------------------------- flash attention
// grid (T/128, B, H); 8 waves = 128 query rows; 32-key chunks via WMMA
// v is head-transposed: vt[(b*H+h)*HD + hd][key], key-stride LKVP_, pad zeroed
// K/V chunks staged via async global->LDS copies. OOB K keys (last chunk only)
// leave stale LDS; those score columns are always masked to -1e30.
__global__ __launch_bounds__(256)
void flash_attn(const __bf16* __restrict__ q, const __bf16* __restrict__ k,
                const __bf16* __restrict__ vt, __bf16* __restrict__ y) {
  __shared__ __bf16 Ks[32][72];       // K chunk: [key][hd]
  __shared__ __bf16 Vt[64][40];       // V chunk: [hd][key]
  __shared__ __bf16 Ps[8][16][40];    // per-wave P staging (C-layout -> A-layout)
  const int tid  = threadIdx.x;
  const int wave = tid >> 5, lane = tid & 31;
  const int lr    = lane & 15;
  const int off   = (lane & 16) ? 8 : 0;
  const int rbase = (lane & 16) ? 8 : 0;
  const int tb = blockIdx.x * 128;
  const int b = blockIdx.y, h = blockIdx.z;
  const int t0w = tb + wave * 16;

  const __bf16* qrow = q + ((size_t)(b * T_) + t0w + lr) * C_ + h * HD_;
  const v16bf aq0 = make_frag(qrow + off,      qrow + off + 16);       // hd 0..31
  const v16bf aq1 = make_frag(qrow + 32 + off, qrow + 32 + off + 16);  // hd 32..63

  float mrow[8], lrow[8];
  v8f acc[4];
#pragma unroll
  for (int i = 0; i < 8; ++i) { mrow[i] = -1e30f; lrow[i] = 0.f; }
#pragma unroll
  for (int ni = 0; ni < 4; ++ni)
#pragma unroll
    for (int i = 0; i < 8; ++i) acc[ni][i] = 0.f;

  const size_t kbase = ((size_t)b * LKV_) * C_ + h * HD_;
  const int krow = tid >> 3;           // 0..31 (key within chunk)
  const int koff = (tid & 7) * 8;      // 0..56 (hd offset)
  const int vhd  = tid >> 2;           // 0..63 (hd row of transposed V)
  const int vks  = (tid & 3) * 8;      // 0..24 (key seg)
  const __bf16* kpg = k + kbase + (size_t)krow * C_ + koff;
  const __bf16* vtb = vt + (((size_t)b * H_ + h) * HD_ + vhd) * LKVP_ + vks;
  __bf16* ldsK = &Ks[krow][koff];
  __bf16* ldsV = &Vt[vhd][vks];
  const int ncols  = tb + 130;         // max visible col (+mem cols, +NU block)
  const int nchunk = (ncols + 31) >> 5;

  for (int ch = 0; ch < nchunk; ++ch) {
    const int c0 = ch * 32;
    __syncthreads();                              // previous chunk consumed
    if (c0 + krow < LKV_) async_copy16(ldsK, kpg + (size_t)c0 * C_);
    async_copy16(ldsV, vtb + c0);                 // pad keys are zeros
    async_wait0();
    __syncthreads();

    const bool active = (c0 <= t0w + 17) || (t0w < NU_ && c0 < 130);
    if (active) {
      v8f s[2];
#pragma unroll
      for (int half = 0; half < 2; ++half) {
        const __bf16* kp = &Ks[half * 16 + lr][0];
        v16bf bk0 = make_frag(kp + off,      kp + off + 16);
        v16bf bk1 = make_frag(kp + 32 + off, kp + 32 + off + 16);
        v8f t;
#pragma unroll
        for (int i = 0; i < 8; ++i) t[i] = 0.f;
        t = wmma_bf16(aq0, bk0, t);
        t = wmma_bf16(aq1, bk1, t);
#pragma unroll
        for (int i = 0; i < 8; ++i) {
          const int trow = t0w + rbase + i;
          const int cc = c0 + half * 16 + lr;
          const bool vis = (cc < 2) || ((cc - 2) <= trow) ||
                           (trow < NU_ && (cc - 2) < NU_);
          s[half][i] = vis ? t[i] * SCALE_ : -1e30f;
        }
      }
      // online softmax; C/D layout rows live across 16-lane groups
#pragma unroll
      for (int i = 0; i < 8; ++i) {
        float mx = fmaxf(s[0][i], s[1][i]);
#pragma unroll
        for (int d = 1; d < 16; d <<= 1) mx = fmaxf(mx, __shfl_xor(mx, d, 16));
        const float mn   = fmaxf(mrow[i], mx);
        const float corr = __expf(mrow[i] - mn);
        const float e0 = __expf(s[0][i] - mn);
        const float e1 = __expf(s[1][i] - mn);
        float rs = e0 + e1;
#pragma unroll
        for (int d = 1; d < 16; d <<= 1) rs += __shfl_xor(rs, d, 16);
        lrow[i] = lrow[i] * corr + rs;
        mrow[i] = mn;
#pragma unroll
        for (int ni = 0; ni < 4; ++ni) acc[ni][i] *= corr;
        Ps[wave][rbase + i][lr]      = f2bf(e0);
        Ps[wave][rbase + i][lr + 16] = f2bf(e1);
      }
      const __bf16* pp = &Ps[wave][lr][0];
      const v16bf ap = make_frag(pp + off, pp + off + 16);
#pragma unroll
      for (int ni = 0; ni < 4; ++ni) {
        const __bf16* vp = &Vt[ni * 16 + lr][0];
        v16bf bv = make_frag(vp + off, vp + off + 16);
        acc[ni] = wmma_bf16(ap, bv, acc[ni]);
      }
    }
  }

#pragma unroll
  for (int ni = 0; ni < 4; ++ni)
#pragma unroll
    for (int i = 0; i < 8; ++i) {
      const int trow = t0w + rbase + i;
      const float inv = 1.f / lrow[i];
      y[((size_t)(b * T_) + trow) * C_ + h * HD_ + ni * 16 + lr] =
          f2bf(acc[ni][i] * inv);
    }
}

// --------------------------------------------------------------------- host
extern "C" void kernel_launch(void* const* d_in, const int* in_sizes, int n_in,
                              void* d_out, int out_size, void* d_ws, size_t ws_size,
                              hipStream_t stream) {
  const float* x      = (const float*)d_in[0];
  const float* prev_m = (const float*)d_in[1];
  const float* cached = (const float*)d_in[2];
  const float* Wq  = (const float*)d_in[3];  const float* bq  = (const float*)d_in[4];
  const float* Wk  = (const float*)d_in[5];  const float* bk  = (const float*)d_in[6];
  const float* Wv  = (const float*)d_in[7];  const float* bv  = (const float*)d_in[8];
  const float* Wp  = (const float*)d_in[9];  const float* bp  = (const float*)d_in[10];
  const float* Wqm = (const float*)d_in[11]; const float* bqm = (const float*)d_in[12];
  const float* Wkm = (const float*)d_in[13]; const float* bkm = (const float*)d_in[14];
  float* out = (float*)d_out;

  char* ws = (char*)d_ws;
  size_t o = 0;
  auto alloc = [&](size_t bytes) -> void* {
    void* p = ws + o; o += (bytes + 255) & ~(size_t)255; return p;
  };
  __bf16* wqT = (__bf16*)alloc((size_t)C_ * C_ * 2);   // transposed [N][K]
  __bf16* wkT = (__bf16*)alloc((size_t)C_ * C_ * 2);
  __bf16* wvT = (__bf16*)alloc((size_t)C_ * C_ * 2);
  __bf16* wpT = (__bf16*)alloc((size_t)C_ * C_ * 2);
  float*  qm_f = (float*)alloc((size_t)B_ * C_ * 4);
  float*  km_f = (float*)alloc((size_t)B_ * M_ * C_ * 4);
  float*  mem_f= (float*)alloc((size_t)B_ * C_ * 4);
  __bf16* cm_b = (__bf16*)alloc((size_t)B_ * LKV_ * C_ * 2);
  __bf16* q_b  = (__bf16*)alloc((size_t)B_ * T_ * C_ * 2);
  __bf16* k_b  = (__bf16*)alloc((size_t)B_ * LKV_ * C_ * 2);
  const size_t vt_bytes = (size_t)B_ * H_ * HD_ * LKVP_ * 2;
  __bf16* vt   = (__bf16*)alloc(vt_bytes);             // head-transposed V
  __bf16* y_b  = (__bf16*)alloc((size_t)B_ * T_ * C_ * 2);
  (void)ws_size; (void)in_sizes; (void)n_in; (void)out_size;

  // weights: transpose+convert to bf16 [N][K]
  transpose_cvt<<<dim3(32, 32), 256, 0, stream>>>(Wq, wqT);
  transpose_cvt<<<dim3(32, 32), 256, 0, stream>>>(Wk, wkT);
  transpose_cvt<<<dim3(32, 32), 256, 0, stream>>>(Wv, wvT);
  transpose_cvt<<<dim3(32, 32), 256, 0, stream>>>(Wp, wpT);

  // memory-compression attention (fp32, tiny)
  row_proj<<<dim3(C_ / 256, B_),      256, 0, stream>>>(prev_m, Wqm, bqm, qm_f);
  row_proj<<<dim3(C_ / 256, B_ * M_), 256, 0, stream>>>(cached, Wkm, bkm, km_f);
  mem_attn<<<(B_ * H_) / 8, 256, 0, stream>>>(qm_f, km_f, mem_f);

  const size_t ncm = (size_t)B_ * LKV_ * C_;
  build_cm<<<(unsigned)((ncm + 255) / 256), 256, 0, stream>>>(x, prev_m, mem_f, cm_b);

  // zero V pad keys once so attention's padded vector loads see exact zeros
  hipMemsetAsync(vt, 0, vt_bytes, stream);

  // QKV projections (bf16 WMMA GEMMs); q reads cm with +2 row offset (x rows)
  gemm_bf16<<<dim3(8, 8, B_), 256, 0, stream>>>(cm_b, LKV_, 2, wqT, bq, q_b, T_,   T_,   1);
  gemm_bf16<<<dim3(9, 8, B_), 256, 0, stream>>>(cm_b, LKV_, 0, wkT, bk, k_b, LKV_, LKV_, 1);
  gemm_bf16<<<dim3(9, 8, B_), 256, 0, stream>>>(cm_b, LKV_, 0, wvT, bv, vt,  LKV_, LKV_, 2);

  flash_attn<<<dim3(T_ / 128, B_, H_), 256, 0, stream>>>(q_b, k_b, vt, y_b);

  // output projection -> fp32 d_out
  gemm_bf16<<<dim3((B_ * T_) / 128, 8, 1), 256, 0, stream>>>(
      y_b, B_ * T_, 0, wpT, bp, out, B_ * T_, B_ * T_, 0);
}
